// Attention_MoE_layer_75136157876593
// MI455X (gfx1250) — matrix operand order repaired and checked by reading, not compiled, not run on hardware
//
#include <hip/hip_runtime.h>

// ---------------------------------------------------------------------------
// MI455X (gfx1250) transformer block: RMSNorm -> MHA -> +res -> RMSNorm ->
// top2/8 MoE -> +res.
// All contractions via v_wmma_f32_16x16x32_bf16 (wave32); shared tiles staged
// through LDS with GLOBAL_LOAD_ASYNC_TO_LDS_B128 (ASYNCcnt double buffering).
// ---------------------------------------------------------------------------

typedef __bf16 bf16_t;
typedef __attribute__((ext_vector_type(16))) __bf16 v16bf;
typedef __attribute__((ext_vector_type(8)))  float  v8f;

#define TQ 8192   // B*S tokens
#define DQ 512
#define SQ 2048
#define BQ 4
#define HQ 8
#define EQ 8
#define FQ 1024

union FragU { v16bf v; uint4 q[2]; };

__device__ __forceinline__ v8f wmma_bf16(v16bf a, v16bf b, v8f c) {
  return __builtin_amdgcn_wmma_f32_16x16x32_bf16(false, a, false, b, (short)0, c,
                                                 false, false);
}

// A-fragment: 16x32 bf16, row-major source, lane (r = lane&15) holds row M=r.
// K elements per lane: [8*hl, 8*hl+8) and [16+8*hl, 16+8*hl+8)  (hl = lane>>4)
__device__ __forceinline__ v16bf load_frag_a(const bf16_t* base, int ld) {
  int lane = threadIdx.x & 31;
  int hl = lane >> 4, r = lane & 15;
  const bf16_t* p = base + (size_t)r * ld + hl * 8;
  FragU f;
  f.q[0] = *reinterpret_cast<const uint4*>(p);
  f.q[1] = *reinterpret_cast<const uint4*>(p + 16);
  return f.v;
}

// A-fragment where each lane already holds its own (gathered) row pointer.
__device__ __forceinline__ v16bf load_frag_a_row(const bf16_t* rowptr, int k0) {
  int lane = threadIdx.x & 31;
  int hl = lane >> 4;
  const bf16_t* p = rowptr + k0 + hl * 8;
  FragU f;
  f.q[0] = *reinterpret_cast<const uint4*>(p);
  f.q[1] = *reinterpret_cast<const uint4*>(p + 16);
  return f.v;
}

// B-fragment: 32x16 bf16 loaded from B^T (rows = N).  Lane n holds column
// N=n&15, K elements [8*hl, 8*hl+8) and [16+8*hl, 16+8*hl+8).
__device__ __forceinline__ v16bf load_frag_bt(const bf16_t* baseT, int ld) {
  int lane = threadIdx.x & 31;
  int hl = lane >> 4, n = lane & 15;
  const bf16_t* p = baseT + (size_t)n * ld + hl * 8;
  FragU f;
  f.q[0] = *reinterpret_cast<const uint4*>(p);
  f.q[1] = *reinterpret_cast<const uint4*>(p + 16);
  return f.v;
}

// ---------------------------------------------------------------------------
// Async global->LDS staging (CDNA5 ASYNCcnt path)
// ---------------------------------------------------------------------------
__device__ __forceinline__ void async_ld16(unsigned lds_addr, const bf16_t* g) {
  asm volatile("global_load_async_to_lds_b128 %0, %1, off"
               :: "v"(lds_addr), "v"(g) : "memory");
}

// Stage a ROWS x COLS bf16 row-major tile (ld elements) into LDS (contiguous
// row-major, row stride COLS).  Block-cooperative: 128 threads, constant trip
// count (no EXEC masking): each thread issues ROWS*COLS/(8*128) async copies.
template <int ROWS, int COLS>
__device__ __forceinline__ void stage_tile(bf16_t* lds, const bf16_t* g, int ld) {
  constexpr int CHUNKS = ROWS * COLS / 8;  // 16-byte chunks
  constexpr int CPR = COLS / 8;            // chunks per row (128 % CPR == 0)
  constexpr int ITER = CHUNKS / 128;       // per-thread async issues
  int c = threadIdx.x;
  int row = c / CPR, ko = (c % CPR) * 8;
  const bf16_t* gp = g + (size_t)row * ld + ko;
  unsigned lp = (unsigned)(size_t)(void*)lds + (unsigned)c * 16;
#pragma unroll
  for (int i = 0; i < ITER; ++i) {
    async_ld16(lp, gp);
    gp += (size_t)(128 / CPR) * ld;
    lp += 128 * 16;
  }
}

// ---------------------------------------------------------------------------
// Shared GEMM core: per-wave 16(M) x 64(N) tile, B (64x32 per K-step) staged
// through LDS with async double buffering, shared by the 4 waves of a block.
//   arow : per-lane A row pointer (lane&15 selects M row; gather-friendly)
//   bt   : B^T base at (n0, 0), row-major, ldb elements per N-row
// ---------------------------------------------------------------------------
template <int KTOT>
__device__ __forceinline__ void gemm_core(const bf16_t* arow, const bf16_t* bt,
                                          int ldb, bf16_t* ldsb, v8f acc[4]) {
  constexpr int NSTEP = KTOT / 32;
  stage_tile<64, 32>(ldsb, bt, ldb);
#pragma unroll 2
  for (int ks = 0; ks < NSTEP; ++ks) {
    bf16_t* cur = ldsb + (ks & 1) * (64 * 32);
    if (ks + 1 < NSTEP) {
      stage_tile<64, 32>(ldsb + ((ks + 1) & 1) * (64 * 32), bt + (ks + 1) * 32, ldb);
      asm volatile("s_wait_asynccnt 2" ::: "memory");
    } else {
      asm volatile("s_wait_asynccnt 0" ::: "memory");
    }
    __syncthreads();
    __builtin_prefetch(arow + ks * 32 + 128, 0, 3);
    v16bf a = load_frag_a_row(arow, ks * 32);
#pragma unroll
    for (int c = 0; c < 4; ++c) {
      v16bf b = load_frag_bt(cur + (size_t)(c * 16) * 32, 32);
      acc[c] = wmma_bf16(a, b, acc[c]);
    }
    __syncthreads();
  }
}

// ---------------------------------------------------------------------------
// Weight conversion: src f32 [batch][R][C] -> dst bf16 [batch][C][R]
// ---------------------------------------------------------------------------
__global__ void __launch_bounds__(256)
transpose_to_bf16_kernel(const float* __restrict__ src, bf16_t* __restrict__ dst,
                         int R, int C, long total) {
  long i = (long)blockIdx.x * 256 + threadIdx.x;
  if (i >= total) return;
  long rc = (long)R * C;
  long bidx = i / rc;
  long rem = i - bidx * rc;
  long r = rem / C, c = rem - r * C;
  dst[bidx * rc + c * (long)R + r] = (bf16_t)src[i];
}

// ---------------------------------------------------------------------------
// RMSNorm (wave per token) -> bf16
// ---------------------------------------------------------------------------
__global__ void __launch_bounds__(256)
rmsnorm_kernel(const float* __restrict__ x, const float* __restrict__ w,
               bf16_t* __restrict__ out) {
  int wave = threadIdx.x >> 5, lane = threadIdx.x & 31;
  int t = blockIdx.x * 8 + wave;
  if (t >= TQ) return;
  const float4* xr = (const float4*)(x + (size_t)t * DQ + lane * 16);
  float xv[16];
  for (int q = 0; q < 4; ++q) {
    float4 v = xr[q];
    xv[q * 4 + 0] = v.x; xv[q * 4 + 1] = v.y;
    xv[q * 4 + 2] = v.z; xv[q * 4 + 3] = v.w;
  }
  float ss = 0.f;
  for (int i = 0; i < 16; ++i) ss += xv[i] * xv[i];
  for (int o = 16; o > 0; o >>= 1) ss += __shfl_xor(ss, o, 32);
  float r = rsqrtf(ss * (1.f / DQ) + 1e-6f);
  const float* wr = w + lane * 16;
  bf16_t* orow = out + (size_t)t * DQ + lane * 16;
  for (int i = 0; i < 16; ++i) orow[i] = (bf16_t)(xv[i] * r * wr[i]);
}

// ---------------------------------------------------------------------------
// Fused QKV GEMM: [T,512] x [512,1536] (+bias).  Writes Q,K natural row-major
// and V transposed as vt[b][h][hd][s].   grid (T/64, 1536/64), block 128.
// ---------------------------------------------------------------------------
__global__ void __launch_bounds__(128)
qkv_gemm_kernel(const bf16_t* __restrict__ xn, const bf16_t* __restrict__ wt,
                const float* __restrict__ bqv, const float* __restrict__ bkv,
                const float* __restrict__ bvv, bf16_t* __restrict__ qb,
                bf16_t* __restrict__ kb, bf16_t* __restrict__ vtb) {
  __shared__ __align__(16) bf16_t bsm[2][64 * 32];
  int wave = threadIdx.x >> 5, lane = threadIdx.x & 31;
  int hl = lane >> 4, n = lane & 15;
  int m0 = blockIdx.x * 64 + wave * 16;
  int n0 = blockIdx.y * 64;
  v8f acc[4] = {};
  const bf16_t* arow = xn + (size_t)(m0 + n) * DQ;
  gemm_core<DQ>(arow, wt + (size_t)n0 * DQ, DQ, bsm[0], acc);
#pragma unroll
  for (int c = 0; c < 4; ++c) {
    int col = n0 + c * 16 + n;
    float bias = (col < 512) ? bqv[col]
               : (col < 1024) ? bkv[col - 512] : bvv[col - 1024];
#pragma unroll
    for (int j = 0; j < 8; ++j) {
      int row = m0 + j + 8 * hl;
      float val = acc[c][j] + bias;
      if (col < 512) {
        qb[(size_t)row * DQ + col] = (bf16_t)val;
      } else if (col < 1024) {
        kb[(size_t)row * DQ + (col - 512)] = (bf16_t)val;
      } else {
        int ch = col - 1024, hh = ch >> 6, hd = ch & 63;
        int bb = row >> 11, ss = row & (SQ - 1);
        vtb[(((size_t)(bb * HQ + hh) * 64 + hd) << 11) + ss] = (bf16_t)val;
      }
    }
  }
}

// ---------------------------------------------------------------------------
// Flash attention, computed transposed (scores^T = K x Q^T) so per-query
// softmax stats stay in-lane.  K and V^T tiles are shared by all 4 waves of a
// block and staged through LDS with async double buffering.
// grid (B*H, S/64), block 128 (wave per 16-query tile).
// ---------------------------------------------------------------------------
__global__ void __launch_bounds__(128)
attn_kernel(const bf16_t* __restrict__ qm, const bf16_t* __restrict__ km,
            const bf16_t* __restrict__ vt, bf16_t* __restrict__ ctxb) {
  __shared__ __align__(16) bf16_t kts[2][32 * 64];   // 32 keys x 64 hd
  __shared__ __align__(16) bf16_t vts[2][64 * 32];   // 64 hd x 32 keys
  __shared__ __align__(16) bf16_t pbuf[4][16 * 32];  // per-wave P^T tile
  int bh = blockIdx.x;
  int b = bh >> 3, h = bh & 7;
  int wave = threadIdx.x >> 5, lane = threadIdx.x & 31;
  int hl = lane >> 4, n = lane & 15;
  int s0 = (blockIdx.y * 4 + wave) * 16;
  const bf16_t* qbase = qm + ((size_t)(b * SQ + s0)) * DQ + h * 64;
  const bf16_t* kbase = km + ((size_t)b * SQ) * DQ + h * 64;
  const bf16_t* vbase = vt + ((size_t)(b * HQ + h) * 64) * (size_t)SQ;
  bf16_t* pw = pbuf[wave];

  // Q B-fragments are loop-invariant (hd chunks 0..31, 32..63)
  v16bf bq0 = load_frag_bt(qbase, DQ);
  v16bf bq1 = load_frag_bt(qbase + 32, DQ);

  v8f acc[4] = {};
  float mrun = -1e30f, lrun = 0.f;

  stage_tile<32, 64>(kts[0], kbase, DQ);
  stage_tile<64, 32>(vts[0], vbase, SQ);
#pragma unroll 2
  for (int ks = 0; ks < SQ / 32; ++ks) {
    const bf16_t* kc = kts[ks & 1];
    const bf16_t* vc = vts[ks & 1];
    if (ks + 1 < SQ / 32) {
      stage_tile<32, 64>(kts[(ks + 1) & 1], kbase + (size_t)(ks + 1) * 32 * DQ, DQ);
      stage_tile<64, 32>(vts[(ks + 1) & 1], vbase + (ks + 1) * 32, SQ);
      asm volatile("s_wait_asynccnt 4" ::: "memory");
    } else {
      asm volatile("s_wait_asynccnt 0" ::: "memory");
    }
    __syncthreads();

    // two 16(keys) x 16(queries) score tiles from the LDS K tile
    v8f s0v = {}, s1v = {};
    s0v = wmma_bf16(load_frag_a(kc, 64), bq0, s0v);
    s0v = wmma_bf16(load_frag_a(kc + 32, 64), bq1, s0v);
    s1v = wmma_bf16(load_frag_a(kc + 16 * 64, 64), bq0, s1v);
    s1v = wmma_bf16(load_frag_a(kc + 16 * 64 + 32, 64), bq1, s1v);

    float tm = -1e30f;
#pragma unroll
    for (int j = 0; j < 8; ++j) {
      s0v[j] *= 0.125f;  // 1/sqrt(HD)
      s1v[j] *= 0.125f;
      tm = fmaxf(tm, fmaxf(s0v[j], s1v[j]));
    }
    tm = fmaxf(tm, __shfl_xor(tm, 16, 32));   // combine lane n with n+16
    float mnew = fmaxf(mrun, tm);
    float alpha = __expf(mrun - mnew);
    float ts = 0.f;
#pragma unroll
    for (int j = 0; j < 8; ++j) {
      float p0 = __expf(s0v[j] - mnew);
      float p1 = __expf(s1v[j] - mnew);
      pw[n * 32 + j + 8 * hl]      = (bf16_t)p0;   // keys 0..15 of step
      pw[n * 32 + 16 + j + 8 * hl] = (bf16_t)p1;   // keys 16..31 of step
      ts += p0 + p1;
    }
    ts += __shfl_xor(ts, 16, 32);
    lrun = lrun * alpha + ts;
    mrun = mnew;
#pragma unroll
    for (int mt = 0; mt < 4; ++mt)
#pragma unroll
      for (int j = 0; j < 8; ++j) acc[mt][j] *= alpha;

    asm volatile("s_wait_dscnt 0" ::: "memory");
    v16bf pb = load_frag_bt(pw, 32);   // B = P (32 keys x 16 queries)
#pragma unroll
    for (int mt = 0; mt < 4; ++mt)
      acc[mt] = wmma_bf16(load_frag_a(vc + (size_t)(mt * 16) * 32, 32), pb, acc[mt]);

    __syncthreads();
  }
  float inv = 1.f / lrun;
  bf16_t* crow = ctxb + (size_t)(b * SQ + s0 + n) * DQ + h * 64;
#pragma unroll
  for (int mt = 0; mt < 4; ++mt)
#pragma unroll
    for (int j = 0; j < 8; ++j)
      crow[mt * 16 + j + 8 * hl] = (bf16_t)(acc[mt][j] * inv);
}

// ---------------------------------------------------------------------------
// Output projection + bias + residual(inputs) -> x1 (f32)
// grid (T/64, 512/64), block 128
// ---------------------------------------------------------------------------
__global__ void __launch_bounds__(128)
outproj_gemm_kernel(const bf16_t* __restrict__ ctxb, const bf16_t* __restrict__ wot,
                    const float* __restrict__ bo, const float* __restrict__ resid,
                    float* __restrict__ x1) {
  __shared__ __align__(16) bf16_t bsm[2][64 * 32];
  int wave = threadIdx.x >> 5, lane = threadIdx.x & 31;
  int hl = lane >> 4, n = lane & 15;
  int m0 = blockIdx.x * 64 + wave * 16;
  int n0 = blockIdx.y * 64;
  v8f acc[4] = {};
  const bf16_t* arow = ctxb + (size_t)(m0 + n) * DQ;
  gemm_core<DQ>(arow, wot + (size_t)n0 * DQ, DQ, bsm[0], acc);
#pragma unroll
  for (int c = 0; c < 4; ++c) {
    int col = n0 + c * 16 + n;
    float bias = bo[col];
#pragma unroll
    for (int j = 0; j < 8; ++j) {
      int row = m0 + j + 8 * hl;
      x1[(size_t)row * DQ + col] = acc[c][j] + bias + resid[(size_t)row * DQ + col];
    }
  }
}

// ---------------------------------------------------------------------------
// RMSNorm2 + gate softmax + top-2 routing (wave per token)
// ---------------------------------------------------------------------------
__global__ void __launch_bounds__(256)
rms2_gate_kernel(const float* __restrict__ x1, const float* __restrict__ w,
                 const float* __restrict__ gw, bf16_t* __restrict__ xn2,
                 int* __restrict__ topi, float* __restrict__ topw) {
  int wave = threadIdx.x >> 5, lane = threadIdx.x & 31;
  int t = blockIdx.x * 8 + wave;
  if (t >= TQ) return;
  const float4* xr = (const float4*)(x1 + (size_t)t * DQ + lane * 16);
  float xv[16];
  for (int q = 0; q < 4; ++q) {
    float4 v = xr[q];
    xv[q * 4 + 0] = v.x; xv[q * 4 + 1] = v.y;
    xv[q * 4 + 2] = v.z; xv[q * 4 + 3] = v.w;
  }
  float ss = 0.f;
  for (int i = 0; i < 16; ++i) ss += xv[i] * xv[i];
  for (int o = 16; o > 0; o >>= 1) ss += __shfl_xor(ss, o, 32);
  float r = rsqrtf(ss * (1.f / DQ) + 1e-6f);
  int d0 = lane * 16;
  const float* wr = w + d0;
  bf16_t* orow = xn2 + (size_t)t * DQ + d0;
  float xs[16];
  for (int i = 0; i < 16; ++i) {
    float xn = xv[i] * r * wr[i];
    xs[i] = xn;
    orow[i] = (bf16_t)xn;
  }
  float lg[EQ];
  for (int e = 0; e < EQ; ++e) {
    float p = 0.f;
    for (int i = 0; i < 16; ++i) p += xs[i] * gw[(size_t)(d0 + i) * EQ + e];
    lg[e] = p;
  }
  for (int o = 16; o > 0; o >>= 1)
    for (int e = 0; e < EQ; ++e) lg[e] += __shfl_xor(lg[e], o, 32);
  if (lane == 0) {
    float mx = lg[0];
    for (int e = 1; e < EQ; ++e) mx = fmaxf(mx, lg[e]);
    float pe[EQ];
    for (int e = 0; e < EQ; ++e) pe[e] = __expf(lg[e] - mx);
    int i0 = 0; float v0 = pe[0];
    for (int e = 1; e < EQ; ++e) if (pe[e] > v0) { v0 = pe[e]; i0 = e; }
    int i1 = (i0 == 0) ? 1 : 0; float v1 = pe[i1];
    for (int e = 0; e < EQ; ++e)
      if (e != i0 && pe[e] > v1) { v1 = pe[e]; i1 = e; }
    float inv = 1.f / (v0 + v1);
    topi[2 * t]     = i0; topw[2 * t]     = v0 * inv;
    topi[2 * t + 1] = i1; topw[2 * t + 1] = v1 * inv;
  }
}

// ---------------------------------------------------------------------------
// Routing helpers
// ---------------------------------------------------------------------------
__global__ void init_counts_kernel(int* counts, int* cursor) {
  int i = threadIdx.x;
  if (i < EQ) { counts[i] = 0; cursor[i] = 0; }
}

__global__ void __launch_bounds__(256)
count_kernel(const int* __restrict__ topi, int* __restrict__ counts) {
  int i = blockIdx.x * 256 + threadIdx.x;
  if (i < TQ * 2) atomicAdd(&counts[topi[i]], 1);
}

__global__ void offsets_kernel(const int* counts, int* offsets) {
  if (threadIdx.x == 0) {
    int run = 0;
    for (int e = 0; e < EQ; ++e) { offsets[e] = run; run += counts[e]; }
  }
}

__global__ void __launch_bounds__(256)
scatter_kernel(const int* __restrict__ topi, const int* __restrict__ offsets,
               int* __restrict__ cursor, int* __restrict__ list) {
  int i = blockIdx.x * 256 + threadIdx.x;
  if (i < TQ * 2) {
    int e = topi[i];
    int p = atomicAdd(&cursor[e], 1);
    list[offsets[e] + p] = i;  // encodes (token<<1)|k
  }
}

__global__ void __launch_bounds__(256)
copy_kernel(const float* __restrict__ src, float* __restrict__ dst) {
  int i = blockIdx.x * 256 + threadIdx.x;
  dst[i] = src[i];
}

// ---------------------------------------------------------------------------
// MoE GEMM 1: h = relu(gather(xn2) @ ew1[e] + eb1[e])   (stores bf16 to hbuf)
// grid (T/64, F/64, E), block 128.  Block-uniform early exit (barriers!).
// ---------------------------------------------------------------------------
__global__ void __launch_bounds__(128)
moe_gemm1_kernel(const bf16_t* __restrict__ xn2, const bf16_t* __restrict__ ew1t,
                 const float* __restrict__ eb1, const int* __restrict__ counts,
                 const int* __restrict__ offsets, const int* __restrict__ list,
                 bf16_t* __restrict__ hbuf) {
  __shared__ __align__(16) bf16_t bsm[2][64 * 32];
  int e = blockIdx.z;
  int cnt = counts[e];
  if ((int)blockIdx.x * 64 >= cnt) return;   // block-uniform
  int wave = threadIdx.x >> 5, lane = threadIdx.x & 31;
  int hl = lane >> 4, n = lane & 15;
  int m0 = blockIdx.x * 64 + wave * 16;
  int base = offsets[e];
  int ridx = m0 + n; if (ridx >= cnt) ridx = cnt - 1;
  const bf16_t* arow = xn2 + (size_t)(list[base + ridx] >> 1) * DQ;
  int n0 = blockIdx.y * 64;
  v8f acc[4] = {};
  gemm_core<DQ>(arow, ew1t + (size_t)e * FQ * DQ + (size_t)n0 * DQ, DQ, bsm[0], acc);
#pragma unroll
  for (int c = 0; c < 4; ++c) {
    int col = n0 + c * 16 + n;
    float bias = eb1[e * FQ + col];
#pragma unroll
    for (int j = 0; j < 8; ++j) {
      int row = m0 + j + 8 * hl;
      if (row < cnt) {
        float h = acc[c][j] + bias;
        hbuf[(size_t)(base + row) * FQ + col] = (bf16_t)(h > 0.f ? h : 0.f);
      }
    }
  }
}

// ---------------------------------------------------------------------------
// MoE GEMM 2: out[t] += w * (hbuf @ ew2[e] + eb2[e])
// grid (T/64, D/64, E), block 128.  Block-uniform early exit.
// ---------------------------------------------------------------------------
__global__ void __launch_bounds__(128)
moe_gemm2_kernel(const bf16_t* __restrict__ hbuf, const bf16_t* __restrict__ ew2t,
                 const float* __restrict__ eb2, const int* __restrict__ counts,
                 const int* __restrict__ offsets, const int* __restrict__ list,
                 const float* __restrict__ topw, float* __restrict__ out) {
  __shared__ __align__(16) bf16_t bsm[2][64 * 32];
  int e = blockIdx.z;
  int cnt = counts[e];
  if ((int)blockIdx.x * 64 >= cnt) return;   // block-uniform
  int wave = threadIdx.x >> 5, lane = threadIdx.x & 31;
  int hl = lane >> 4, n = lane & 15;
  int m0 = blockIdx.x * 64 + wave * 16;
  int base = offsets[e];
  int rr = m0 + n; if (rr >= cnt) rr = cnt - 1;
  const bf16_t* arow = hbuf + (size_t)(base + rr) * FQ;
  int n0 = blockIdx.y * 64;
  v8f acc[4] = {};
  gemm_core<FQ>(arow, ew2t + (size_t)e * DQ * FQ + (size_t)n0 * FQ, FQ, bsm[0], acc);
#pragma unroll
  for (int j = 0; j < 8; ++j) {
    int row = m0 + j + 8 * hl;
    if (row < cnt) {
      int li = list[base + row];
      int t = li >> 1;
      float wgt = topw[li];
#pragma unroll
      for (int c = 0; c < 4; ++c) {
        int col = n0 + c * 16 + n;
        float val = (acc[c][j] + eb2[e * DQ + col]) * wgt;
#if defined(__AMDGCN__)
        unsafeAtomicAdd(&out[(size_t)t * DQ + col], val);
#else
        atomicAdd(&out[(size_t)t * DQ + col], val);
#endif
      }
    }
  }
}

// ---------------------------------------------------------------------------
// Host-side orchestration
// ---------------------------------------------------------------------------
extern "C" void kernel_launch(void* const* d_in, const int* in_sizes, int n_in,
                              void* d_out, int out_size, void* d_ws, size_t ws_size,
                              hipStream_t stream) {
  const float* inputs = (const float*)d_in[0];
  const float* rms1_w = (const float*)d_in[1];
  const float* wq     = (const float*)d_in[2];
  const float* bq     = (const float*)d_in[3];
  const float* wk     = (const float*)d_in[4];
  const float* bk     = (const float*)d_in[5];
  const float* wv     = (const float*)d_in[6];
  const float* bv     = (const float*)d_in[7];
  const float* wo     = (const float*)d_in[8];
  const float* bo     = (const float*)d_in[9];
  const float* rms2_w = (const float*)d_in[10];
  const float* gate_w = (const float*)d_in[11];
  const float* ew1    = (const float*)d_in[12];
  const float* eb1    = (const float*)d_in[13];
  const float* ew2    = (const float*)d_in[14];
  const float* eb2    = (const float*)d_in[15];
  float* out = (float*)d_out;

  char* ws = (char*)d_ws;
  size_t off = 0;
  auto carve = [&](size_t bytes) -> char* {
    char* p = ws + off;
    off = (off + bytes + 255) & ~(size_t)255;
    return p;
  };
  bf16_t* xn1   = (bf16_t*)carve((size_t)TQ * DQ * 2);
  bf16_t* qb    = (bf16_t*)carve((size_t)TQ * DQ * 2);
  bf16_t* kb    = (bf16_t*)carve((size_t)TQ * DQ * 2);
  bf16_t* vtb   = (bf16_t*)carve((size_t)TQ * DQ * 2);
  bf16_t* ctxb  = (bf16_t*)carve((size_t)TQ * DQ * 2);
  float*  x1    = (float*) carve((size_t)TQ * DQ * 4);
  bf16_t* xn2   = (bf16_t*)carve((size_t)TQ * DQ * 2);
  bf16_t* wqkvt = (bf16_t*)carve((size_t)1536 * DQ * 2);
  bf16_t* wot   = (bf16_t*)carve((size_t)DQ * DQ * 2);
  bf16_t* ew1t  = (bf16_t*)carve((size_t)EQ * DQ * FQ * 2);
  bf16_t* ew2t  = (bf16_t*)carve((size_t)EQ * FQ * DQ * 2);
  int*    topi  = (int*)   carve((size_t)TQ * 2 * 4);
  float*  topw  = (float*) carve((size_t)TQ * 2 * 4);
  int*    counts  = (int*) carve(64);
  int*    offsets = (int*) carve(64);
  int*    cursor  = (int*) carve(64);
  int*    list    = (int*) carve((size_t)TQ * 2 * 4);
  bf16_t* hbuf  = (bf16_t*)carve((size_t)TQ * 2 * FQ * 2);

  // 1) weight transposes -> bf16 B^T layouts
  long wsz = (long)DQ * DQ;
  transpose_to_bf16_kernel<<<(wsz + 255) / 256, 256, 0, stream>>>(wq, wqkvt, DQ, DQ, wsz);
  transpose_to_bf16_kernel<<<(wsz + 255) / 256, 256, 0, stream>>>(wk, wqkvt + (size_t)512 * DQ, DQ, DQ, wsz);
  transpose_to_bf16_kernel<<<(wsz + 255) / 256, 256, 0, stream>>>(wv, wqkvt + (size_t)1024 * DQ, DQ, DQ, wsz);
  transpose_to_bf16_kernel<<<(wsz + 255) / 256, 256, 0, stream>>>(wo, wot, DQ, DQ, wsz);
  long e1sz = (long)EQ * DQ * FQ;
  transpose_to_bf16_kernel<<<(e1sz + 255) / 256, 256, 0, stream>>>(ew1, ew1t, DQ, FQ, e1sz);
  transpose_to_bf16_kernel<<<(e1sz + 255) / 256, 256, 0, stream>>>(ew2, ew2t, FQ, DQ, e1sz);

  // 2) RMSNorm1
  rmsnorm_kernel<<<TQ / 8, 256, 0, stream>>>(inputs, rms1_w, xn1);

  // 3) fused QKV GEMM
  qkv_gemm_kernel<<<dim3(TQ / 64, 1536 / 64), 128, 0, stream>>>(
      xn1, wqkvt, bq, bk, bv, qb, kb, vtb);

  // 4) flash attention
  attn_kernel<<<dim3(BQ * HQ, SQ / 64), 128, 0, stream>>>(qb, kb, vtb, ctxb);

  // 5) output projection + residual -> x1
  outproj_gemm_kernel<<<dim3(TQ / 64, DQ / 64), 128, 0, stream>>>(
      ctxb, wot, bo, inputs, x1);

  // 6) RMSNorm2 + gate top-2
  rms2_gate_kernel<<<TQ / 8, 256, 0, stream>>>(x1, rms2_w, gate_w, xn2, topi, topw);

  // 7) routing
  init_counts_kernel<<<1, 32, 0, stream>>>(counts, cursor);
  count_kernel<<<(TQ * 2) / 256, 256, 0, stream>>>(topi, counts);
  offsets_kernel<<<1, 32, 0, stream>>>(counts, offsets);
  scatter_kernel<<<(TQ * 2) / 256, 256, 0, stream>>>(topi, offsets, cursor, list);

  // 8) out = x1 (residual base)
  copy_kernel<<<(TQ * DQ) / 256, 256, 0, stream>>>(x1, out);

  // 9) expert GEMMs (sparse top-2)
  moe_gemm1_kernel<<<dim3(TQ / 64, FQ / 64, EQ), 128, 0, stream>>>(
      xn2, ew1t, eb1, counts, offsets, list, hbuf);
  moe_gemm2_kernel<<<dim3(TQ / 64, DQ / 64, EQ), 128, 0, stream>>>(
      hbuf, ew2t, eb2, counts, offsets, list, topw, out);

  (void)in_sizes; (void)n_in; (void)out_size; (void)ws_size;
}
